// TransformerBlock_17652315587330
// MI455X (gfx1250) — matrix operand order repaired and checked by reading, not compiled, not run on hardware
//
#include <hip/hip_runtime.h>

// ---------------------------------------------------------------------------
// Transformer block for MI455X (gfx1250, wave32, WMMA).
// All matmuls run through v_wmma_f32_16x16x32_bf16 (f32 accumulate).
// GEMM: 64x128 block tile, 4 waves in a 2x2 grid, 32x64 per wave so each
// A/B tile is reused by 2 waves and each wave does 8 WMMAs per 12 b128 loads.
// ---------------------------------------------------------------------------

typedef __attribute__((ext_vector_type(16))) __bf16 v16bf;
typedef __attribute__((ext_vector_type(8)))  float  v8f;

constexpr int B_  = 2;
constexpr int S_  = 2048;
constexpr int E_  = 1024;
constexpr int H_  = 16;
constexpr int FF_ = 4096;
constexpr int HD_ = E_ / H_;   // 64
constexpr int M_  = B_ * S_;   // 4096
constexpr float EPS_ = 1e-6f;

#define DEVFN static __device__ __forceinline__

DEVFN unsigned short f2bf(float f) {
  union { float f; unsigned u; } x{f};
  unsigned r = x.u + 0x7fffu + ((x.u >> 16) & 1u);
  return (unsigned short)(r >> 16);
}

union TileU { v16bf v; uint4 q[2]; };

// A-operand tile (16x32, MxK), ISA 7.12.2 16-bit A layout:
// lanes 0-15: row=lane, V0-3=K0..7, V4-7=K16..23; lanes 16-31: K8..15 / K24..31
DEVFN v16bf load_a(const unsigned short* base, int ld, int lane) {
  int l = lane & 15, hi = (lane >> 4) & 1;
  const unsigned short* p = base + (size_t)l * ld + hi * 8;
  TileU u;
  u.q[0] = *reinterpret_cast<const uint4*>(p);
  u.q[1] = *reinterpret_cast<const uint4*>(p + 16);
  return u.v;
}

// B-operand tile (32x16, KxN), 16-bit B layout matching A:
// lane = column; lanes 0-15: V0-3=K0..7, V4-7=K16..23; lanes 16-31: K8..15 / K24..31
DEVFN v16bf load_b(const unsigned short* base, int ld, int lane) {
  int l = lane & 15, hi = (lane >> 4) & 1;
  const unsigned short* p = base + (size_t)l * ld + hi * 8;
  TileU u;
  u.q[0] = *reinterpret_cast<const uint4*>(p);
  u.q[1] = *reinterpret_cast<const uint4*>(p + 16);
  return u.v;
}

DEVFN v8f wmma_bf16(v16bf a, v16bf b, v8f c) {
  return __builtin_amdgcn_wmma_f32_16x16x32_bf16(false, a, false, b, (short)0, c,
                                                 false, false);
}

// ---------------------------------------------------------------------------
// f32 -> bf16 convert
// ---------------------------------------------------------------------------
__global__ void cvt_f32_bf16(const float* __restrict__ in,
                             unsigned short* __restrict__ out, int n) {
  int i = blockIdx.x * blockDim.x + threadIdx.x;
  if (i < n) out[i] = f2bf(in[i]);
}

// ---------------------------------------------------------------------------
// Generic bf16 GEMM:  C[M,N] = A[M,K] * B[N,K]^T  (+ bias, optional ReLU)
// Block = 4 waves as 2x2 over a 64x128 tile; each wave owns 32 rows x 64 cols
// (2 A-tiles x 4 B-tiles -> 8 WMMAs per K-step). Whole working set stays in
// the 192MB L2, so tile reuse comes from L0/L2 without LDS staging.
// ---------------------------------------------------------------------------
__global__ __launch_bounds__(128) void gemm_bf16(
    const unsigned short* __restrict__ A, const unsigned short* __restrict__ Bm,
    float* Cf, unsigned short* Cb, const float* __restrict__ bias,
    int Ndim, int Kdim, int relu) {
  int lane = threadIdx.x & 31;
  int wave = threadIdx.x >> 5;
  int row0 = blockIdx.y * 64 + (wave >> 1) * 32;
  int col0 = blockIdx.x * 128 + (wave & 1) * 64;

  v8f acc[2][4] = {};
  const unsigned short* ar0 = A + (size_t)row0 * Kdim;
  const unsigned short* ar1 = A + (size_t)(row0 + 16) * Kdim;
  for (int k = 0; k < Kdim; k += 32) {
    v16bf a0 = load_a(ar0 + k, Kdim, lane);
    v16bf a1 = load_a(ar1 + k, Kdim, lane);
#pragma unroll
    for (int t = 0; t < 4; ++t) {
      v16bf b = load_b(Bm + (size_t)(col0 + 16 * t) * Kdim + k, Kdim, lane);
      acc[0][t] = wmma_bf16(a0, b, acc[0][t]);
      acc[1][t] = wmma_bf16(a1, b, acc[1][t]);
    }
  }

  int l = lane & 15, hi = (lane >> 4) & 1;
#pragma unroll
  for (int rt = 0; rt < 2; ++rt) {
#pragma unroll
    for (int t = 0; t < 4; ++t) {
      int col = col0 + 16 * t + l;
      float bv = bias ? bias[col] : 0.0f;
#pragma unroll
      for (int r = 0; r < 8; ++r) {
        int row = row0 + rt * 16 + r + hi * 8;   // C layout: lanes16-31 -> M=r+8
        float v = acc[rt][t][r] + bv;
        if (relu) v = fmaxf(v, 0.0f);
        size_t idx = (size_t)row * Ndim + col;
        if (Cf) Cf[idx] = v;
        else    Cb[idx] = f2bf(v);
      }
    }
  }
}

// ---------------------------------------------------------------------------
// Build V^T [b,h,hd,s] from kqv [b*s, 3E] so V is K-contiguous for P*V WMMA.
// ---------------------------------------------------------------------------
__global__ void make_vT(const unsigned short* __restrict__ kqv,
                        unsigned short* __restrict__ vT) {
  size_t i = (size_t)blockIdx.x * blockDim.x + threadIdx.x;
  size_t n = (size_t)B_ * H_ * HD_ * S_;
  if (i >= n) return;
  int s  = (int)(i % S_);
  size_t t = i / S_;
  int hd = (int)(t % HD_);
  int bh = (int)(t / HD_);
  int h  = bh & (H_ - 1);
  int b  = bh >> 4;
  vT[i] = kqv[((size_t)(b * S_ + s)) * (3 * E_) + 2 * E_ + h * HD_ + hd];
}

// ---------------------------------------------------------------------------
// Flash attention: one wave per 16-query tile. QK^T via WMMA (HD=64 -> 2 ops),
// causal mask + online softmax with half-wave shuffles, P transposed through
// LDS and fed back as WMMA A-operand for P*V.
// ---------------------------------------------------------------------------
__global__ __launch_bounds__(128) void attn_kernel(
    const unsigned short* __restrict__ kqv, const unsigned short* __restrict__ vT,
    unsigned short* __restrict__ out) {
  __shared__ __align__(16) unsigned short plds[4][16][32];

  int lane = threadIdx.x & 31;
  int wave = threadIdx.x >> 5;
  int tile = blockIdx.x * 4 + wave;
  int qt = tile & (S_ / 16 - 1);     // S_/16 == 128
  int bh = tile >> 7;
  int h = bh & (H_ - 1);
  int b = bh >> 4;
  int q0 = qt * 16;
  int l = lane & 15, hi = (lane >> 4) & 1;

  // Q tile (A-operand), hd chunks 0..31 and 32..63
  const unsigned short* qbase =
      kqv + ((size_t)(b * S_ + q0)) * (3 * E_) + E_ + h * HD_;
  v16bf qa0 = load_a(qbase, 3 * E_, lane);
  v16bf qa1 = load_a(qbase + 32, 3 * E_, lane);

  v8f o0 = {}, o1 = {}, o2 = {}, o3 = {};
  float mrow[8], lrow[8];
#pragma unroll
  for (int r = 0; r < 8; ++r) { mrow[r] = -__builtin_inff(); lrow[r] = 0.0f; }

  int nIter = (q0 + 16 + 31) / 32;   // causal: keys 0 .. q0+15
  for (int j = 0; j < nIter; ++j) {
    int k0 = j * 32;
    // scores for 32 keys = two 16x16 WMMA tiles
    v8f s0 = {}, s1 = {};
    {
      const unsigned short* kb =
          kqv + (size_t)(b * S_ + k0) * (3 * E_) + h * HD_;
      s0 = wmma_bf16(qa0, load_b(kb, 3 * E_, lane), s0);
      s0 = wmma_bf16(qa1, load_b(kb + 32, 3 * E_, lane), s0);
      const unsigned short* kb2 = kb + (size_t)16 * (3 * E_);
      s1 = wmma_bf16(qa0, load_b(kb2, 3 * E_, lane), s1);
      s1 = wmma_bf16(qa1, load_b(kb2 + 32, 3 * E_, lane), s1);
    }
    // scale + causal mask + per-block row max
    float bmax[8];
#pragma unroll
    for (int r = 0; r < 8; ++r) {
      int q = q0 + r + hi * 8;
      int key = k0 + l;
      float v0 = (key      <= q) ? s0[r] * 0.125f : -__builtin_inff();
      float v1 = (key + 16 <= q) ? s1[r] * 0.125f : -__builtin_inff();
      s0[r] = v0; s1[r] = v1;
      bmax[r] = fmaxf(v0, v1);
    }
#pragma unroll
    for (int r = 0; r < 8; ++r) {      // reduce across the 16-lane half
      float m = bmax[r];
      m = fmaxf(m, __shfl_xor(m, 1));
      m = fmaxf(m, __shfl_xor(m, 2));
      m = fmaxf(m, __shfl_xor(m, 4));
      m = fmaxf(m, __shfl_xor(m, 8));
      bmax[r] = m;
    }
    float alpha[8], psum[8];
#pragma unroll
    for (int r = 0; r < 8; ++r) {
      float nm = fmaxf(mrow[r], bmax[r]);
      alpha[r] = expf(mrow[r] - nm);   // first iter: exp(-inf)=0
      mrow[r] = nm;
    }
#pragma unroll
    for (int r = 0; r < 8; ++r) {
      float p0 = expf(s0[r] - mrow[r]);
      float p1 = expf(s1[r] - mrow[r]);
      psum[r] = p0 + p1;
      plds[wave][r + hi * 8][l]      = f2bf(p0);  // transpose P via LDS
      plds[wave][r + hi * 8][16 + l] = f2bf(p1);
    }
#pragma unroll
    for (int r = 0; r < 8; ++r) {
      float sm = psum[r];
      sm += __shfl_xor(sm, 1);
      sm += __shfl_xor(sm, 2);
      sm += __shfl_xor(sm, 4);
      sm += __shfl_xor(sm, 8);
      lrow[r] = lrow[r] * alpha[r] + sm;
      o0[r] *= alpha[r]; o1[r] *= alpha[r];
      o2[r] *= alpha[r]; o3[r] *= alpha[r];
    }
    // P (16x32) back from LDS as A-operand; V from vT as B-operand
    v16bf pa = load_a(&plds[wave][0][0], 32, lane);
    const unsigned short* vb = vT + ((size_t)bh * HD_) * S_ + k0;
    o0 = wmma_bf16(pa, load_b(vb + (size_t) 0 * S_, S_, lane), o0);
    o1 = wmma_bf16(pa, load_b(vb + (size_t)16 * S_, S_, lane), o1);
    o2 = wmma_bf16(pa, load_b(vb + (size_t)32 * S_, S_, lane), o2);
    o3 = wmma_bf16(pa, load_b(vb + (size_t)48 * S_, S_, lane), o3);
  }

  v8f os[4] = {o0, o1, o2, o3};
#pragma unroll
  for (int t = 0; t < 4; ++t) {
#pragma unroll
    for (int r = 0; r < 8; ++r) {
      int q  = q0 + r + hi * 8;
      int hd = t * 16 + l;
      out[((size_t)(b * S_ + q)) * E_ + h * HD_ + hd] =
          f2bf(os[t][r] / lrow[r]);
    }
  }
}

// ---------------------------------------------------------------------------
// Fused residual + LayerNorm: out = LN(a + res) * gamma + beta
// One block per row of E_ elements; optional bf16 copy of the output.
// ---------------------------------------------------------------------------
__global__ __launch_bounds__(256) void ln_fused(
    const float* __restrict__ a, const float* __restrict__ res,
    const float* __restrict__ gamma, const float* __restrict__ beta,
    float* __restrict__ outf, unsigned short* outb) {
  __shared__ float row[E_];
  __shared__ float red[256];
  int r = blockIdx.x, t = threadIdx.x;
  const float* pa = a + (size_t)r * E_;
  const float* pb = res + (size_t)r * E_;

  float s = 0.0f;
  for (int i = t; i < E_; i += 256) {
    float v = pa[i] + pb[i];
    row[i] = v;
    s += v;
  }
  red[t] = s;
  __syncthreads();
  for (int off = 128; off > 0; off >>= 1) {
    if (t < off) red[t] += red[t + off];
    __syncthreads();
  }
  float mu = red[0] * (1.0f / E_);
  __syncthreads();

  s = 0.0f;
  for (int i = t; i < E_; i += 256) {
    float d = row[i] - mu;
    s += d * d;
  }
  red[t] = s;
  __syncthreads();
  for (int off = 128; off > 0; off >>= 1) {
    if (t < off) red[t] += red[t + off];
    __syncthreads();
  }
  float rstd = rsqrtf(red[0] * (1.0f / E_) + EPS_);

  for (int i = t; i < E_; i += 256) {
    float v = (row[i] - mu) * rstd * gamma[i] + beta[i];
    outf[(size_t)r * E_ + i] = v;
    if (outb) outb[(size_t)r * E_ + i] = f2bf(v);
  }
}

// ---------------------------------------------------------------------------
// Host-side orchestration
// ---------------------------------------------------------------------------
extern "C" void kernel_launch(void* const* d_in, const int* in_sizes, int n_in,
                              void* d_out, int out_size, void* d_ws,
                              size_t ws_size, hipStream_t stream) {
  (void)in_sizes; (void)n_in; (void)out_size; (void)ws_size;

  const float* x       = (const float*)d_in[0];
  // d_in[1] = causal mask (implicit in attention kernel)
  const float* w_kqv   = (const float*)d_in[2];
  const float* w_o     = (const float*)d_in[3];
  const float* w_up    = (const float*)d_in[4];
  const float* bias_up = (const float*)d_in[5];
  const float* w_down  = (const float*)d_in[6];
  const float* bias_dn = (const float*)d_in[7];
  const float* g1 = (const float*)d_in[8];
  const float* b1 = (const float*)d_in[9];
  const float* g2 = (const float*)d_in[10];
  const float* b2 = (const float*)d_in[11];
  float* out = (float*)d_out;
  char* ws = (char*)d_ws;

  constexpr size_t MiB = 1ull << 20;
  unsigned short* xb     = (unsigned short*)(ws + 0);          //  8 MiB
  unsigned short* wkqvb  = (unsigned short*)(ws + 8  * MiB);   //  6 MiB
  unsigned short* wob    = (unsigned short*)(ws + 14 * MiB);   //  2 MiB
  unsigned short* wupb   = (unsigned short*)(ws + 16 * MiB);   //  8 MiB
  unsigned short* wdownb = (unsigned short*)(ws + 24 * MiB);   //  8 MiB
  unsigned short* kqvb   = (unsigned short*)(ws + 32 * MiB);   // 24 MiB
  unsigned short* vT     = (unsigned short*)(ws + 56 * MiB);   //  8 MiB
  unsigned short* attnb  = (unsigned short*)(ws + 64 * MiB);   //  8 MiB
  float*          proj   = (float*)(ws + 72 * MiB);            // 16 MiB
  float*          x1     = (float*)(ws + 88 * MiB);            // 16 MiB
  unsigned short* x1b    = (unsigned short*)(ws + 104 * MiB);  //  8 MiB
  unsigned short* hb     = (unsigned short*)(ws + 32 * MiB);   // 32 MiB (reuse kqv+vT)
  float*          tmp    = (float*)(ws + 64 * MiB);            // 16 MiB (reuse attn+proj)

  auto cvt = [&](const float* src, unsigned short* dst, int n) {
    cvt_f32_bf16<<<(n + 255) / 256, 256, 0, stream>>>(src, dst, n);
  };
  cvt(x,      xb,     M_ * E_);
  cvt(w_kqv,  wkqvb,  3 * E_ * E_);
  cvt(w_o,    wob,    E_ * E_);
  cvt(w_up,   wupb,   FF_ * E_);
  cvt(w_down, wdownb, E_ * FF_);

  dim3 blk(128);
  // kqv = x @ w_kqv^T  (4096 x 3072, K=1024) -> bf16
  gemm_bf16<<<dim3(3 * E_ / 128, M_ / 64), blk, 0, stream>>>(
      xb, wkqvb, nullptr, kqvb, nullptr, 3 * E_, E_, 0);
  {
    int n = B_ * H_ * HD_ * S_;
    make_vT<<<(n + 255) / 256, 256, 0, stream>>>(kqvb, vT);
  }
  attn_kernel<<<B_ * H_ * (S_ / 16) / 4, blk, 0, stream>>>(kqvb, vT, attnb);
  // proj = attn @ w_o^T  (4096 x 1024, K=1024) -> f32
  gemm_bf16<<<dim3(E_ / 128, M_ / 64), blk, 0, stream>>>(
      attnb, wob, proj, nullptr, nullptr, E_, E_, 0);
  ln_fused<<<M_, 256, 0, stream>>>(proj, x, g1, b1, x1, x1b);
  // h = relu(x1 @ w_up^T + b_up)  (4096 x 4096, K=1024) -> bf16
  gemm_bf16<<<dim3(FF_ / 128, M_ / 64), blk, 0, stream>>>(
      x1b, wupb, nullptr, hb, bias_up, FF_, E_, 1);
  // tmp = h @ w_down^T + b_down  (4096 x 1024, K=4096) -> f32
  gemm_bf16<<<dim3(E_ / 128, M_ / 64), blk, 0, stream>>>(
      hb, wdownb, tmp, nullptr, bias_dn, E_, FF_, 0);
  ln_fused<<<M_, 256, 0, stream>>>(tmp, x1, g2, b2, out, nullptr);
}
